// MomentumLSTM_83854941487585
// MI455X (gfx1250) — matrix operand order repaired and checked by reading, not compiled, not run on hardware
//
#include <hip/hip_runtime.h>

typedef __attribute__((ext_vector_type(16))) _Float16 v16h;
typedef __attribute__((ext_vector_type(8)))  _Float16 v8h;
typedef __attribute__((ext_vector_type(8)))  float    v8f;

#define B_TOTAL 32768
#define T_STEPS 60
#define D_IN 7
#define H1 64
#define H2 32
#define WAVES_PER_BLOCK 4
#define BLOCK_THREADS (WAVES_PER_BLOCK * 32)
#define ROWS_PER_WAVE 16

// LDS layout, units of _Float16 (halves). All offsets multiples of 512 halves.
#define OFF_WIH1 0        // 16 fragments (Ktot=32: K<7 weights, K==7 bias1)
#define OFF_WHH1 8192     // 32 fragments (Ktot=64)
#define OFF_WIH2 24576    // 24 fragments (Ktot=96: K<64 weights, K==64 bias2)
#define OFF_WHH2 36864    // 8 fragments  (Ktot=32)
#define OFF_WD   40960    // 2 fragments  (Ktot=64: K<32 weights, K==32 bd)
#define OFF_STAGE 41984   // per-wave: 1536 (h1 16x96) + 1024 (h2 16x64)
#define STAGE_PER_WAVE 2560
#define SMEM_HALVES (OFF_STAGE + WAVES_PER_BLOCK * STAGE_PER_WAVE)

__device__ __forceinline__ v8f wmma16(v16h a, v16h b, v8f c) {
  return __builtin_amdgcn_wmma_f32_16x16x32_f16(false, a, false, b, (short)0, c,
                                                false, false);
}

__device__ __forceinline__ float sigm(float x) {
  return 1.0f / (1.0f + __expf(-x));
}
__device__ __forceinline__ float tanh_fast(float x) {
  float e = __expf(-2.0f * x);
  return (1.0f - e) / (1.0f + e);
}

// Convert W (row-major [Ntot x Kact] f32) into f16 B-matrix fragments for
// V_WMMA_*_16x16x32. fragment index = kc*nfragN + nt; within a fragment:
// lane*16 + half, where K = kc*32 + (lane>=16 ? 16 : 0) + half,
// N = nt*16 + (lane & 15). Row K==Kact (if ba != null) holds ba[N](+bb[N]),
// folding the bias into the GEMM via a 1.0 in the matching A column.
__device__ void swizzleB(const float* __restrict__ W,
                         const float* __restrict__ ba,
                         const float* __restrict__ bb, _Float16* dst,
                         int nfragN, int Ktot, int Kact, int tid) {
  int total = (Ktot / 32) * nfragN * 512;
  for (int e = tid; e < total; e += BLOCK_THREADS) {
    int frag = e >> 9;
    int r = e & 511;
    int lane = r >> 4;
    int hh = r & 15;
    int kc = frag / nfragN;
    int nt = frag - kc * nfragN;
    int K = kc * 32 + ((lane >> 4) << 4) + hh;
    int N = nt * 16 + (lane & 15);
    float v = 0.0f;
    if (K < Kact) {
      v = W[N * Kact + K];
    } else if (K == Kact && ba != nullptr) {
      v = ba[N];
      if (bb != nullptr) v += bb[N];
    }
    dst[e] = (_Float16)v;
  }
}

__device__ __forceinline__ v16h bfrag(const _Float16* base, int idx, int lane) {
  return *(const v16h*)(base + idx * 512 + lane * 16);
}

// Load an A-matrix f16 fragment (16x32, K-chunk kk) from a row-major
// [16 x stride] f16 staging tile. Lane l<16: M=l, halves K=kk*32+0..15;
// lane>=16: M=l-16, halves K=kk*32+16..31.
__device__ __forceinline__ v16h afrag(const _Float16* stage, int stride, int kk,
                                      int lane) {
  int m = lane & 15;
  int ko = (lane >> 4) << 4;
  const _Float16* p = stage + m * stride + kk * 32 + ko;
  v8h lo = *(const v8h*)(p);
  v8h hi = *(const v8h*)(p + 8);
  v16h a;
#pragma unroll
  for (int q = 0; q < 8; ++q) { a[q] = lo[q]; a[q + 8] = hi[q]; }
  return a;
}

__global__ void __launch_bounds__(BLOCK_THREADS)
MomentumLSTM_fused_kernel(const float* __restrict__ x,
                          const float* __restrict__ Wih1,
                          const float* __restrict__ Whh1,
                          const float* __restrict__ bih1,
                          const float* __restrict__ bhh1,
                          const float* __restrict__ Wih2,
                          const float* __restrict__ Whh2,
                          const float* __restrict__ bih2,
                          const float* __restrict__ bhh2,
                          const float* __restrict__ Wd,
                          const float* __restrict__ bd,
                          const float* __restrict__ Wo,
                          const float* __restrict__ bo,
                          float* __restrict__ out) {
  extern __shared__ _Float16 smem[];
  _Float16* s_wih1 = smem + OFF_WIH1;
  _Float16* s_whh1 = smem + OFF_WHH1;
  _Float16* s_wih2 = smem + OFF_WIH2;
  _Float16* s_whh2 = smem + OFF_WHH2;
  _Float16* s_wd = smem + OFF_WD;

  const int tid = threadIdx.x;
  const int lane = tid & 31;
  const int wave = tid >> 5;
  const int nl = lane & 15;
  _Float16* h1s = smem + OFF_STAGE + wave * STAGE_PER_WAVE;  // 16 x 96
  _Float16* h2s = h1s + 1536;                                // 16 x 64

  // One-time weight swizzle into WMMA B-fragment layout (f32 -> f16),
  // with biases folded into the K==Kact row.
  swizzleB(Wih1, bih1, bhh1, s_wih1, 16, 32, D_IN, tid);  // [256 x 7]+bias1
  swizzleB(Whh1, nullptr, nullptr, s_whh1, 16, 64, H1, tid);
  swizzleB(Wih2, bih2, bhh2, s_wih2, 8, 96, H1, tid);     // [128 x 64]+bias2
  swizzleB(Whh2, nullptr, nullptr, s_whh2, 8, 32, H2, tid);
  swizzleB(Wd, bd, nullptr, s_wd, 1, 64, H2, tid);        // [16 x 32]+bd

  // Constant staging columns: h1 col 64 == 1.0 (picks bias2 row of Wih2),
  // h2 col 32 == 1.0 (picks bd row of Wd). Written once, never touched by
  // the per-timestep stores (which cover cols < 64 / < 32 only).
  for (int e = lane; e < 16 * 32; e += 32) {
    int m = e >> 5;
    int cc = e & 31;
    h1s[m * 96 + 64 + cc] = (cc == 0) ? (_Float16)1.0f : (_Float16)0.0f;
    h2s[m * 64 + 32 + cc] = (cc == 0) ? (_Float16)1.0f : (_Float16)0.0f;
  }
  __syncthreads();

  const v16h h1fb = afrag(h1s, 96, 2, lane);  // constant bias-selector A frag
  const v16h h2fb = afrag(h2s, 64, 1, lane);  // constant bias-selector A frag

  const int row0 = (blockIdx.x * WAVES_PER_BLOCK + wave) * ROWS_PER_WAVE;
  const int mh = (lane >> 4) << 3;  // C/D layout: row = q + mh

  v8f c1[4];
  v8f c2[2];
#pragma unroll
  for (int j = 0; j < 4; ++j) c1[j] = {};
#pragma unroll
  for (int j = 0; j < 2; ++j) c2[j] = {};
  v16h h1f0 = {}, h1f1 = {}, h2f0 = {};
  const v8f vz = {};

#pragma unroll 1
  for (int t = 0; t < T_STEPS; ++t) {
    // ---- x A-fragment: K=0..6 data, K=7 == 1.0 (bias1 selector), lanes 0..15
    v16h xa = {};
    if (lane < 16) {
      const float* xp =
          x + (size_t)(row0 + lane) * (T_STEPS * D_IN) + t * D_IN;
#pragma unroll
      for (int d = 0; d < D_IN; ++d) xa[d] = (_Float16)xp[d];
      xa[7] = (_Float16)1.0f;
      if (t + 1 < T_STEPS) __builtin_prefetch(xp + D_IN, 0, 1);
    }

    // ---- layer 1: gates[16x256], 4 aligned tiles (i,f,g,o) at a time ----
#pragma unroll
    for (int j = 0; j < 4; ++j) {
      v8f gi, gf, gg, go;
      // input projection + bias (K=32 covers padded D=7 plus bias row)
      gi = wmma16(xa, bfrag(s_wih1, j, lane), vz);
      gf = wmma16(xa, bfrag(s_wih1, j + 4, lane), vz);
      gg = wmma16(xa, bfrag(s_wih1, j + 8, lane), vz);
      go = wmma16(xa, bfrag(s_wih1, j + 12, lane), vz);
      // recurrent h1 @ Whh1^T (K=64 -> 2 chunks)
      gi = wmma16(h1f0, bfrag(s_whh1, j, lane), gi);
      gi = wmma16(h1f1, bfrag(s_whh1, 16 + j, lane), gi);
      gf = wmma16(h1f0, bfrag(s_whh1, j + 4, lane), gf);
      gf = wmma16(h1f1, bfrag(s_whh1, 16 + j + 4, lane), gf);
      gg = wmma16(h1f0, bfrag(s_whh1, j + 8, lane), gg);
      gg = wmma16(h1f1, bfrag(s_whh1, 16 + j + 8, lane), gg);
      go = wmma16(h1f0, bfrag(s_whh1, j + 12, lane), go);
      go = wmma16(h1f1, bfrag(s_whh1, 16 + j + 12, lane), go);
#pragma unroll
      for (int q = 0; q < 8; ++q) {
        float iv = sigm(gi[q]);
        float fv = sigm(gf[q]);
        float gv = tanh_fast(gg[q]);
        float ov = sigm(go[q]);
        float c = fv * c1[j][q] + iv * gv;
        c1[j][q] = c;
        float h = ov * tanh_fast(c);
        h1s[(q + mh) * 96 + j * 16 + nl] = (_Float16)h;
      }
    }
    __syncthreads();
    h1f0 = afrag(h1s, 96, 0, lane);
    h1f1 = afrag(h1s, 96, 1, lane);

    // ---- layer 2: gates[16x128], input = h1(t), state = h2(t-1) ----
#pragma unroll
    for (int j = 0; j < 2; ++j) {
      v8f gi, gf, gg, go;
      // bias chunk first (constant selector A-fragment picks K==64 bias row)
      gi = wmma16(h1fb, bfrag(s_wih2, 16 + j, lane), vz);
      gf = wmma16(h1fb, bfrag(s_wih2, 16 + j + 2, lane), vz);
      gg = wmma16(h1fb, bfrag(s_wih2, 16 + j + 4, lane), vz);
      go = wmma16(h1fb, bfrag(s_wih2, 16 + j + 6, lane), vz);
      // input projection h1 @ Wih2^T (K=64 -> 2 chunks)
      gi = wmma16(h1f0, bfrag(s_wih2, j, lane), gi);
      gi = wmma16(h1f1, bfrag(s_wih2, 8 + j, lane), gi);
      gf = wmma16(h1f0, bfrag(s_wih2, j + 2, lane), gf);
      gf = wmma16(h1f1, bfrag(s_wih2, 8 + j + 2, lane), gf);
      gg = wmma16(h1f0, bfrag(s_wih2, j + 4, lane), gg);
      gg = wmma16(h1f1, bfrag(s_wih2, 8 + j + 4, lane), gg);
      go = wmma16(h1f0, bfrag(s_wih2, j + 6, lane), go);
      go = wmma16(h1f1, bfrag(s_wih2, 8 + j + 6, lane), go);
      // recurrent h2 @ Whh2^T
      gi = wmma16(h2f0, bfrag(s_whh2, j, lane), gi);
      gf = wmma16(h2f0, bfrag(s_whh2, j + 2, lane), gf);
      gg = wmma16(h2f0, bfrag(s_whh2, j + 4, lane), gg);
      go = wmma16(h2f0, bfrag(s_whh2, j + 6, lane), go);
#pragma unroll
      for (int q = 0; q < 8; ++q) {
        float iv = sigm(gi[q]);
        float fv = sigm(gf[q]);
        float gv = tanh_fast(gg[q]);
        float ov = sigm(go[q]);
        float c = fv * c2[j][q] + iv * gv;
        c2[j][q] = c;
        float h = ov * tanh_fast(c);
        h2s[(q + mh) * 64 + j * 16 + nl] = (_Float16)h;
      }
    }
    __syncthreads();
    h2f0 = afrag(h2s, 64, 0, lane);
  }

  // ---- dense head: relu(h2 @ Wd^T + bd) -> sigmoid(. @ Wo^T + bo) ----
  v8f dacc = wmma16(h2fb, bfrag(s_wd, 1, lane), vz);  // bd via selector
  dacc = wmma16(h2f0, bfrag(s_wd, 0, lane), dacc);

  float wv = Wo[nl];
  float bov = bo[0];
  v8f p;
#pragma unroll
  for (int q = 0; q < 8; ++q) p[q] = fmaxf(dacc[q], 0.0f) * wv;
  // reduce over N (16 lanes per half-wave); rows live in q + mh
#pragma unroll
  for (int q = 0; q < 8; ++q) {
    float s = p[q];
    s += __shfl_xor(s, 1, 32);
    s += __shfl_xor(s, 2, 32);
    s += __shfl_xor(s, 4, 32);
    s += __shfl_xor(s, 8, 32);
    p[q] = s;
  }
  if (nl == 0) {
#pragma unroll
    for (int q = 0; q < 8; ++q)
      out[row0 + mh + q] = sigm(p[q] + bov);
  }
}

extern "C" void kernel_launch(void* const* d_in, const int* in_sizes, int n_in,
                              void* d_out, int out_size, void* d_ws,
                              size_t ws_size, hipStream_t stream) {
  const float* x = (const float*)d_in[0];
  const float* Wih1 = (const float*)d_in[1];
  const float* Whh1 = (const float*)d_in[2];
  const float* bih1 = (const float*)d_in[3];
  const float* bhh1 = (const float*)d_in[4];
  const float* Wih2 = (const float*)d_in[5];
  const float* Whh2 = (const float*)d_in[6];
  const float* bih2 = (const float*)d_in[7];
  const float* bhh2 = (const float*)d_in[8];
  const float* Wd = (const float*)d_in[9];
  const float* bd = (const float*)d_in[10];
  const float* Wo = (const float*)d_in[11];
  const float* bo = (const float*)d_in[12];
  float* out = (float*)d_out;

  dim3 grid(B_TOTAL / (WAVES_PER_BLOCK * ROWS_PER_WAVE));  // 512
  dim3 block(BLOCK_THREADS);                                // 128 (4 waves)
  size_t smem_bytes = (size_t)SMEM_HALVES * sizeof(_Float16);
  MomentumLSTM_fused_kernel<<<grid, block, smem_bytes, stream>>>(
      x, Wih1, Whh1, bih1, bhh1, Wih2, Whh2, bih2, bhh2, Wd, bd, Wo, bo, out);
}
